// TPFp8AmpereLinear_6725918786086
// MI455X (gfx1250) — matrix operand order repaired and checked by reading, not compiled, not run on hardware
//
#include <hip/hip_runtime.h>
#include <stdint.h>

// ---------------------------------------------------------------------------
// Problem dims (fixed by the reference)
// ---------------------------------------------------------------------------
#define MDIM 4096
#define KDIM 4096
#define NDIM 16384
#define RDIM 32

// LDS layout (bank-conflict-free padded strides)
#define A_STRIDE 144            // 128B data + 16B pad  -> lane banks 4*k, distinct
#define B_STRIDE 80             //  64B data + 16B pad  -> lane banks 4*k, distinct
#define A_BUF    (128 * A_STRIDE)   // 18432 B
#define B_BUF    (128 * B_STRIDE)   // 10240 B
#define B_BASE   (2 * A_BUF)        // 36864
#define SMEM_SZ  (2 * A_BUF + 2 * B_BUF)  // 57344 B

typedef __attribute__((ext_vector_type(8)))  __bf16       v8bf;
typedef __attribute__((ext_vector_type(16))) __bf16       v16bf;
typedef __attribute__((ext_vector_type(8)))  float        v8f;
typedef __attribute__((ext_vector_type(2)))  float        v2f;
typedef __attribute__((ext_vector_type(4)))  unsigned int v4u;
typedef __attribute__((ext_vector_type(4)))  float        v4f;
typedef __attribute__((ext_vector_type(4)))  int          v4i;

#if defined(__has_builtin)
#  if __has_builtin(__builtin_amdgcn_cvt_pk_f32_fp8)
#    define USE_HW_FP8_DEC 1
#  endif
#  if __has_builtin(__builtin_amdgcn_global_load_async_to_lds_b128)
#    define USE_ASYNC_LDS 1
#  endif
#  if __has_builtin(__builtin_amdgcn_s_wait_asynccnt)
#    define USE_WAIT_ASYNC_BUILTIN 1
#  endif
#endif

// 16-byte global -> LDS copy. Async (ASYNCcnt-tracked) when the gfx1250
// builtin is available, otherwise a synchronous load+ds_store fallback.
__device__ __forceinline__ void cp16_g2l(const void* g, void* l) {
#if defined(USE_ASYNC_LDS)
  __builtin_amdgcn_global_load_async_to_lds_b128(
      (__attribute__((address_space(1))) v4i*)g,
      (__attribute__((address_space(3))) v4i*)l, 0, 0);
#else
  *(v4u*)l = *(const v4u*)g;
#endif
}

template <int N>
__device__ __forceinline__ void wait_async() {
#if defined(USE_ASYNC_LDS)
#  if defined(USE_WAIT_ASYNC_BUILTIN)
  __builtin_amdgcn_s_wait_asynccnt((short)N);
#  else
  asm volatile("s_wait_asynccnt %0" ::"n"(N) : "memory");
#  endif
#endif
}

// ---------------------------------------------------------------------------
// fp8 e4m3fn helpers
// ---------------------------------------------------------------------------
__device__ __forceinline__ unsigned int f32_to_e4m3(float f) {
  unsigned int u   = __float_as_uint(f);
  unsigned int sgn = (u >> 24) & 0x80u;
  unsigned int a   = u & 0x7fffffffu;
  if (a >= 0x7f800000u) return sgn | 0x7fu;     // inf / nan -> nan
  if (a >= 0x43e40000u) return sgn | 0x7eu;     // >= 456 saturates to 448
  if (a <  0x3c800000u) {                       // < 2^-6 : subnormal region
    float q = __uint_as_float(a) * 512.0f;      // units of 2^-9
    unsigned int qi = (unsigned int)(int)rintf(q);
    return sgn | qi;
  }
  unsigned int m = a + 0x7ffffu + ((a >> 20) & 1u);  // RNE into 3-bit mantissa
  unsigned int e = (m >> 23) - 120u;                 // rebias 127 -> 7
  return sgn | (e << 3) | ((m >> 20) & 7u);
}

__device__ __forceinline__ float e4m3_to_f32_sw(unsigned int b) {
  unsigned short h = (unsigned short)(((b & 0x80u) << 8) | ((b & 0x7fu) << 7));
  _Float16 hf = __builtin_bit_cast(_Float16, h);
  return (float)hf * 256.0f;
}

template <int OFS>
__device__ __forceinline__ void deq4(unsigned int dw, float s, v16bf& o) {
#if defined(USE_HW_FP8_DEC)
  v2f lo = __builtin_amdgcn_cvt_pk_f32_fp8((int)dw, false);
  v2f hi = __builtin_amdgcn_cvt_pk_f32_fp8((int)dw, true);
  o[OFS + 0] = (__bf16)(lo.x * s);
  o[OFS + 1] = (__bf16)(lo.y * s);
  o[OFS + 2] = (__bf16)(hi.x * s);
  o[OFS + 3] = (__bf16)(hi.y * s);
#else
  o[OFS + 0] = (__bf16)(e4m3_to_f32_sw((dw      ) & 0xffu) * s);
  o[OFS + 1] = (__bf16)(e4m3_to_f32_sw((dw >>  8) & 0xffu) * s);
  o[OFS + 2] = (__bf16)(e4m3_to_f32_sw((dw >> 16) & 0xffu) * s);
  o[OFS + 3] = (__bf16)(e4m3_to_f32_sw((dw >> 24) & 0xffu) * s);
#endif
}

// ---------------------------------------------------------------------------
// Kernel 1: quantize fp32 weight [N,K] -> fp8 e4m3 in workspace (cold pass)
// ---------------------------------------------------------------------------
__global__ __launch_bounds__(256) void k_quant(const float* __restrict__ w,
                                               const float* __restrict__ scale,
                                               unsigned char* __restrict__ wq) {
  size_t base = ((size_t)blockIdx.x * 256u + threadIdx.x) * 16u;
  int    n    = (int)(base >> 12);  // / KDIM
  float  s    = scale[n];
  const v4f* wp = (const v4f*)(w + base);
  v4u out;
#pragma unroll
  for (int d = 0; d < 4; ++d) {
    v4f f = wp[d];
    unsigned int r = f32_to_e4m3(f.x / s);
    r |= f32_to_e4m3(f.y / s) << 8;
    r |= f32_to_e4m3(f.z / s) << 16;
    r |= f32_to_e4m3(f.w / s) << 24;
    out[d] = r;
  }
  *(v4u*)(wq + base) = out;
}

// ---------------------------------------------------------------------------
// Kernel 2: t[M,R] = x[M,K] @ lora_A[R,K]^T   (bf16, WMMA, 1 wave / block)
// ---------------------------------------------------------------------------
__global__ __launch_bounds__(32) void k_lora_t(const __bf16* __restrict__ x,
                                               const __bf16* __restrict__ lA,
                                               __bf16* __restrict__ t) {
  const int lane = threadIdx.x & 31;
  const int hf   = lane >> 4;
  const int mr   = lane & 15;
  const int m0   = blockIdx.y * 16;
  const int r0   = blockIdx.x * 16;

  const __bf16* pa = x  + (size_t)(m0 + mr) * KDIM + hf * 8;
  const __bf16* pb = lA + (size_t)(r0 + mr) * KDIM + hf * 8;

  v8f acc = {};
#pragma unroll 2
  for (int kk = 0; kk < KDIM; kk += 32) {
    v8bf a0 = *(const v8bf*)(pa + kk);
    v8bf a1 = *(const v8bf*)(pa + kk + 16);
    v16bf av = __builtin_shufflevector(a0, a1, 0,1,2,3,4,5,6,7,8,9,10,11,12,13,14,15);
    v8bf b0 = *(const v8bf*)(pb + kk);
    v8bf b1 = *(const v8bf*)(pb + kk + 16);
    v16bf bv = __builtin_shufflevector(b0, b1, 0,1,2,3,4,5,6,7,8,9,10,11,12,13,14,15);
    acc = __builtin_amdgcn_wmma_f32_16x16x32_bf16(false, av, false, bv,
                                                  (short)0, acc, false, false);
  }
#pragma unroll
  for (int r = 0; r < 8; ++r)
    t[(size_t)(m0 + hf * 8 + r) * RDIM + r0 + mr] = (__bf16)acc[r];
}

// ---------------------------------------------------------------------------
// Kernel 3: main fp8-dequant GEMM + fused LoRA residual + bias.
//   out[M,N] = x @ deq(wq)^T + t @ lora_B^T + bias
//
// Block = 256 threads (8 waves), block tile 128x128, BK = 64, double-buffered
// LDS fed by gfx1250 async global->LDS copies (ASYNCcnt / s_wait_asynccnt).
// Waves: 2 along M (64 rows) x 4 along N (32 cols); wave tile 64x32 =
// 4(M) x 2(N) WMMA tiles; two K=32 WMMA sub-steps per stage.
//
// LDS rows are padded (A: 144B, B: 80B) so the 16 per-lane b128 reads of a
// fragment start at 16 distinct multiple-of-4 dword banks -> conflict-free.
// ---------------------------------------------------------------------------
__global__ __launch_bounds__(256) void k_gemm(const __bf16* __restrict__ x,
                                              const unsigned char* __restrict__ wq,
                                              const float* __restrict__ scale,
                                              const __bf16* __restrict__ bias,
                                              const __bf16* __restrict__ lB,
                                              const __bf16* __restrict__ t,
                                              __bf16* __restrict__ out) {
  __shared__ __attribute__((aligned(16))) unsigned char smem[SMEM_SZ];

  const int tid  = threadIdx.x;
  const int lane = tid & 31;
  const int wave = tid >> 5;            // 0..7
  const int hf   = lane >> 4;
  const int mr   = lane & 15;

  const size_t blockM = (size_t)blockIdx.y * 128;
  const size_t blockN = (size_t)blockIdx.x * 128;
  const size_t waveM  = blockM + (size_t)(wave >> 2) * 64;   // 64-row slab

  // ---- staging: one K=64 stage copies A(128x128B) + B(128x64B) into LDS.
  // Per thread: 4 A-chunks + 2 B-chunks of 16B => 6 async instrs per wave.
  auto stage = [&](int kk, int buf) {
    unsigned char* sa = smem + buf * A_BUF;
    unsigned char* sb = smem + B_BASE + buf * B_BUF;
#pragma unroll
    for (int p = 0; p < 4; ++p) {
      int c = tid + 256 * p;            // 0..1023
      int row = c >> 3, q = c & 7;      // 8 chunks per row (128B of data)
      cp16_g2l(x + (blockM + row) * (size_t)KDIM + kk + q * 8,
               sa + row * A_STRIDE + q * 16);
    }
#pragma unroll
    for (int p = 0; p < 2; ++p) {
      int c = tid + 256 * p;            // 0..511
      int row = c >> 2, q = c & 3;      // 4 chunks per row (64B of data)
      cp16_g2l(wq + (blockN + row) * (size_t)KDIM + kk + q * 16,
               sb + row * B_STRIDE + q * 16);
    }
  };

  // ---- per-lane LDS fragment offsets (byte offsets inside a buffer)
  int offA[4];
#pragma unroll
  for (int i = 0; i < 4; ++i)
    offA[i] = ((wave >> 2) * 64 + i * 16 + mr) * A_STRIDE + hf * 32;
  int   offB[2];
  float sj[2];
  int   rB[2];
#pragma unroll
  for (int j = 0; j < 2; ++j) {
    int nrow = (wave & 3) * 32 + j * 16 + mr;
    offB[j] = nrow * B_STRIDE + hf * 16;
    rB[j]   = (int)(blockN + nrow);
    sj[j]   = (float)(__bf16)scale[rB[j]];  // reference dequants with bf16 scale
  }

  v8f acc[4][2] = {};

  constexpr int STAGES = KDIM / 64;     // 64
  stage(0, 0);
  for (int s = 0; s < STAGES; ++s) {
    const int cur = s & 1;
    if (s + 1 < STAGES) {
      stage((s + 1) * 64, 1 - cur);     // prefetch next stage (6 async ops)
      wait_async<6>();                  // oldest 6 (stage s) have landed
    } else {
      wait_async<0>();
    }
    __syncthreads();                    // LDS buffer `cur` valid for all waves

    const unsigned char* sa = smem + cur * A_BUF;
    const unsigned char* sb = smem + B_BASE + cur * B_BUF;

#pragma unroll
    for (int ks = 0; ks < 64; ks += 32) {       // two K=32 WMMA sub-steps
      v16bf a[4];
#pragma unroll
      for (int i = 0; i < 4; ++i) {
        const unsigned char* p = sa + offA[i] + ks * 2;
        v8bf a0 = *(const v8bf*)(p);
        v8bf a1 = *(const v8bf*)(p + 16);
        a[i] = __builtin_shufflevector(a0, a1, 0,1,2,3,4,5,6,7,8,9,10,11,12,13,14,15);
      }
      v16bf b[2];
#pragma unroll
      for (int j = 0; j < 2; ++j) {
        v4u q = *(const v4u*)(sb + offB[j] + ks);
        v16bf bv;
        deq4<0>(q.x, sj[j], bv);
        deq4<4>(q.y, sj[j], bv);
        deq4<8>(q.z, sj[j], bv);
        deq4<12>(q.w, sj[j], bv);
        b[j] = bv;
      }
#pragma unroll
      for (int i = 0; i < 4; ++i)
#pragma unroll
        for (int j = 0; j < 2; ++j)
          acc[i][j] = __builtin_amdgcn_wmma_f32_16x16x32_bf16(
              false, a[i], false, b[j], (short)0, acc[i][j], false, false);
    }
    __syncthreads();                    // done reading buffer `cur`
  }

  // ---- fused LoRA residual: one extra K=32 WMMA step (A from t, B from lora_B)
  {
    v16bf al[4];
#pragma unroll
    for (int i = 0; i < 4; ++i) {
      const __bf16* pt = t + (waveM + i * 16 + mr) * (size_t)RDIM + hf * 8;
      v8bf a0 = *(const v8bf*)(pt);
      v8bf a1 = *(const v8bf*)(pt + 16);
      al[i] = __builtin_shufflevector(a0, a1, 0,1,2,3,4,5,6,7,8,9,10,11,12,13,14,15);
    }
    v16bf bl[2];
#pragma unroll
    for (int j = 0; j < 2; ++j) {
      const __bf16* pl = lB + (size_t)rB[j] * RDIM + hf * 8;
      v8bf b0 = *(const v8bf*)(pl);
      v8bf b1 = *(const v8bf*)(pl + 16);
      bl[j] = __builtin_shufflevector(b0, b1, 0,1,2,3,4,5,6,7,8,9,10,11,12,13,14,15);
    }
#pragma unroll
    for (int i = 0; i < 4; ++i)
#pragma unroll
      for (int j = 0; j < 2; ++j)
        acc[i][j] = __builtin_amdgcn_wmma_f32_16x16x32_bf16(
            false, al[i], false, bl[j], (short)0, acc[i][j], false, false);
  }

  // ---- bias + store (C layout: elem r -> M = base + hf*8 + r, N = base + mr)
#pragma unroll
  for (int j = 0; j < 2; ++j) {
    const float bv = (float)bias[rB[j]];
#pragma unroll
    for (int i = 0; i < 4; ++i) {
#pragma unroll
      for (int r = 0; r < 8; ++r) {
        size_t row = waveM + i * 16 + hf * 8 + r;
        out[row * (size_t)NDIM + rB[j]] = (__bf16)(acc[i][j][r] + bv);
      }
    }
  }
}

// ---------------------------------------------------------------------------
// Host-side launch
// ---------------------------------------------------------------------------
extern "C" void kernel_launch(void* const* d_in, const int* in_sizes, int n_in,
                              void* d_out, int out_size, void* d_ws, size_t ws_size,
                              hipStream_t stream) {
  const __bf16* x      = (const __bf16*)d_in[0];
  const float*  weight = (const float*)d_in[1];
  const float*  scale  = (const float*)d_in[2];
  const __bf16* bias   = (const __bf16*)d_in[3];
  const __bf16* lora_A = (const __bf16*)d_in[4];
  const __bf16* lora_B = (const __bf16*)d_in[5];
  __bf16*       out    = (__bf16*)d_out;

  unsigned char* wq = (unsigned char*)d_ws;
  __bf16*        t  = (__bf16*)((char*)d_ws + (size_t)NDIM * KDIM);

  (void)in_sizes; (void)n_in; (void)out_size; (void)ws_size;

  int qblocks = (int)(((size_t)NDIM * KDIM) / (256u * 16u));  // 16384
  k_quant<<<qblocks, 256, 0, stream>>>(weight, scale, wq);

  k_lora_t<<<dim3(RDIM / 16, MDIM / 16), 32, 0, stream>>>(x, lora_A, t);

  k_gemm<<<dim3(NDIM / 128, MDIM / 128), 256, 0, stream>>>(
      x, wq, scale, bias, lora_B, t, out);
}
